// HRM_21431886807185
// MI455X (gfx1250) — compile-verified, matrix operand order checked by reading, not run on hardware
//
#include <hip/hip_runtime.h>
#include <hip/hip_bf16.h>

typedef __attribute__((ext_vector_type(16))) _Float16 v16h;
typedef __attribute__((ext_vector_type(8)))  _Float16 v8h;
typedef __attribute__((ext_vector_type(8)))  float    v8f;
typedef __attribute__((ext_vector_type(4)))  int      v4i;

// exact pointer types for the async-copy builtin (AS1 = global, AS3 = LDS)
typedef __attribute__((address_space(1))) v4i gas_v4i;
typedef __attribute__((address_space(3))) v4i las_v4i;

#define Hd 1024
#define Bd 8
#define Sd 256
#define Vd 32000
#define H3 3072

#ifndef __has_builtin
#define __has_builtin(x) 0
#endif

// ---------------------------------------------------------------------------
// Fragment builder from a (16B-aligned) pointer: 16 halves as two b128 loads.
// ISA 16-bit A/B 16x32 per-lane layout: lanes 0-15 -> row=lane, K={0..7,16..23}
//                                       lanes 16-31 -> row=lane-16, K={8..15,24..31}
// ---------------------------------------------------------------------------
__device__ __forceinline__ v16h load_frag_p(const _Float16* p) {
  v8h lo = *reinterpret_cast<const v8h*>(p);
  v8h hi = *reinterpret_cast<const v8h*>(p + 16);
  v16h f;
#pragma unroll
  for (int i = 0; i < 8; ++i) { f[i] = lo[i]; f[i + 8] = hi[i]; }
  return f;
}

// ---------------------------------------------------------------------------
// WMMA GEMM: C[M,N] = A[M,K] * B[N,K]^T + bias[N]
// block = 256 threads = 8 waves; BN = 128 (16 N-cols per wave), BM = 16*MT.
// A is staged through LDS in KC=256 chunks via ASYNCcnt-tracked
// global_load_async_to_lds_b128 (shared by all 8 waves -> 8x cut in A global
// traffic). B streams direct from L2. LDS pitch padded (+8 halves) so the
// fragment ds_load_b128 is bank-conflict-free.
// ---------------------------------------------------------------------------
template <int MT>
__global__ void gemm_wmma_f16f32(const _Float16* __restrict__ A,
                                 const _Float16* __restrict__ Bm,
                                 const float* __restrict__ bias,
                                 float* __restrict__ C,
                                 int M, int N, int K) {
  constexpr int ROWS = MT * 16;
  constexpr int KC = 256;          // K chunk staged in LDS
  constexpr int AP = KC + 8;       // padded LDS pitch (halves): 528B = 132 dw
  __shared__ _Float16 Atile[ROWS * AP];

  const int tid  = threadIdx.x;
  const int lane = tid & 31;
  const int wave = tid >> 5;
  const int n0 = blockIdx.x * 128 + wave * 16;
  const int m0 = blockIdx.y * ROWS;

  const int fr   = lane & 15;          // fragment row within 16-row tile
  const int kOff = (lane >> 4) << 3;   // 0 or 8

  v8f acc[MT] = {};

  for (int kc = 0; kc < K; kc += KC) {
    __syncthreads();  // previous chunk's readers done before overwrite

    // ---- stage A chunk: ROWS x KC halves, 16B per thread per iteration ----
    const _Float16* src = A + (size_t)m0 * K + kc;
#pragma unroll
    for (int v = tid; v < ROWS * (KC / 8); v += 256) {
      const int row = v >> 5;          // KC/8 == 32 vectors per row
      const int c8  = (v & 31) << 3;
#if __has_builtin(__builtin_amdgcn_global_load_async_to_lds_b128)
      __builtin_amdgcn_global_load_async_to_lds_b128(
          (gas_v4i*)(src + (size_t)row * K + c8),
          (las_v4i*)(Atile + row * AP + c8),
          0, 0);
#else
      *reinterpret_cast<v8h*>(Atile + row * AP + c8) =
          *reinterpret_cast<const v8h*>(src + (size_t)row * K + c8);
#endif
    }
#if __has_builtin(__builtin_amdgcn_global_load_async_to_lds_b128)
#if __has_builtin(__builtin_amdgcn_s_wait_asynccnt)
    __builtin_amdgcn_s_wait_asynccnt(0);
#else
    asm volatile("s_wait_asynccnt 0x0" ::: "memory");
#endif
#endif
    __syncthreads();  // staged data visible to all waves

    // ---- compute on the staged chunk ----
    for (int k0 = 0; k0 < KC; k0 += 32) {
      v16h bf = load_frag_p(Bm + (size_t)(n0 + fr) * K + (kc + k0 + kOff));
#pragma unroll
      for (int t = 0; t < MT; ++t) {
        v16h af = load_frag_p(Atile + (t * 16 + fr) * AP + k0 + kOff);
        acc[t] = __builtin_amdgcn_wmma_f32_16x16x32_f16(
            /*neg_a=*/false, af, /*neg_b=*/false, bf,
            /*c_mod=*/(short)0, acc[t], /*reuse_a=*/false, /*reuse_b=*/false);
      }
    }
  }

  // C layout: VGPR r, lane L -> m = r + 8*(L>=16), n = L&15
  const int cn = n0 + fr;
  const float bv = bias ? bias[cn] : 0.0f;
  const int mh = (lane >> 4) * 8;
#pragma unroll
  for (int t = 0; t < MT; ++t) {
    const int mb = m0 + t * 16 + mh;
#pragma unroll
    for (int r = 0; r < 8; ++r) {
      C[(size_t)(mb + r) * N + cn] = acc[t][r] + bv;
    }
  }
}

// ---------------------------------------------------------------------------
// f32 -> f16 weight conversion (strided source handles the l_Wih split)
// ---------------------------------------------------------------------------
__global__ void cvt_f16(const float* __restrict__ src, _Float16* __restrict__ dst,
                        int rows, int cols, int src_ld) {
  int i = blockIdx.x * blockDim.x + threadIdx.x;
  if (i >= rows * cols) return;
  int r = i / cols, c = i - r * cols;
  dst[(size_t)r * cols + c] = (_Float16)src[(size_t)r * src_ld + c];
}

// Embedding gather -> f16 rows r = b*S + s
__global__ void embed_f16(const int* __restrict__ x, const float* __restrict__ emb,
                          _Float16* __restrict__ xe) {
  int i = blockIdx.x * blockDim.x + threadIdx.x;  // 2048*1024
  int r = i >> 10, k = i & 1023;
  int tok = x[r];
  xe[(size_t)r * Hd + k] = (_Float16)emb[(size_t)tok * Hd + k];
}

// Initialize h_l / h_h f32 state + zero-padded (16-row) f16 copies
__global__ void init_state(const float* __restrict__ hl0, const float* __restrict__ hh0,
                           float* __restrict__ hl, float* __restrict__ hh,
                           _Float16* __restrict__ hlh, _Float16* __restrict__ hhh) {
  int i = blockIdx.x * blockDim.x + threadIdx.x;  // 16*1024
  int row = i >> 10;
  float vl = (row < Bd) ? hl0[i] : 0.0f;
  float vh = (row < Bd) ? hh0[i] : 0.0f;
  hlh[i] = (_Float16)vl;
  hhh[i] = (_Float16)vh;
  if (row < Bd) { hl[i] = vl; hh[i] = vh; }
}

__device__ __forceinline__ float sigmoidf_(float v) {
  return 1.0f / (1.0f + __expf(-v));
}

// Low-level GRU gate math for step s. gix has l_bih folded in; ghl has l_bhh.
__global__ void gru_low_update(const float* __restrict__ gix,   // [2048,3H]
                               const float* __restrict__ gihh,  // [16,3H]
                               const float* __restrict__ ghl,   // [16,3H]
                               float* __restrict__ hl,          // [8,H]
                               _Float16* __restrict__ hlh,      // [16,H]
                               _Float16* __restrict__ hsh,      // [2048,H]
                               int s) {
  int i = blockIdx.x * blockDim.x + threadIdx.x;  // 8*1024
  int b = i >> 10, j = i & 1023;
  size_t row = (size_t)b * Sd + s;
  const float* gx = gix + row * H3;
  const float* gb = gihh + (size_t)b * H3;
  const float* gr = ghl + (size_t)b * H3;
  float r = sigmoidf_(gx[j] + gb[j] + gr[j]);
  float z = sigmoidf_(gx[Hd + j] + gb[Hd + j] + gr[Hd + j]);
  float n = tanhf(gx[2 * Hd + j] + gb[2 * Hd + j] + r * gr[2 * Hd + j]);
  float h = (1.0f - z) * n + z * hl[i];
  hl[i] = h;
  hlh[i] = (_Float16)h;
  hsh[row * Hd + j] = (_Float16)h;
}

// High-level GRU update (once per block of T steps)
__global__ void gru_high_update(const float* __restrict__ gi,  // [16,3H] (+h_bih)
                                const float* __restrict__ gh,  // [16,3H] (+h_bhh)
                                float* __restrict__ hh,
                                _Float16* __restrict__ hhh) {
  int i = blockIdx.x * blockDim.x + threadIdx.x;  // 8*1024
  int b = i >> 10, j = i & 1023;
  const float* gib = gi + (size_t)b * H3;
  const float* ghb = gh + (size_t)b * H3;
  float r = sigmoidf_(gib[j] + ghb[j]);
  float z = sigmoidf_(gib[Hd + j] + ghb[Hd + j]);
  float n = tanhf(gib[2 * Hd + j] + r * ghb[2 * Hd + j]);
  float h = (1.0f - z) * n + z * hh[i];
  hh[i] = h;
  hhh[i] = (_Float16)h;
}

// Final h_l, h_h -> tail of d_out
__global__ void write_states(const float* __restrict__ hl, const float* __restrict__ hh,
                             float* __restrict__ out) {
  int i = blockIdx.x * blockDim.x + threadIdx.x;  // 16384
  out[(size_t)Bd * Sd * Vd + i] = (i < Bd * Hd) ? hl[i] : hh[i - Bd * Hd];
}

// ---------------------------------------------------------------------------
extern "C" void kernel_launch(void* const* d_in, const int* in_sizes, int n_in,
                              void* d_out, int out_size, void* d_ws, size_t ws_size,
                              hipStream_t stream) {
  (void)in_sizes; (void)n_in; (void)out_size; (void)ws_size;
  const int*   x     = (const int*)d_in[0];
  const float* hl0   = (const float*)d_in[1];
  const float* hh0   = (const float*)d_in[2];
  const float* emb   = (const float*)d_in[3];
  const float* l_Wih = (const float*)d_in[4];   // [3H, 2H]
  const float* l_Whh = (const float*)d_in[5];   // [3H, H]
  const float* l_bih = (const float*)d_in[6];
  const float* l_bhh = (const float*)d_in[7];
  const float* h_Wih = (const float*)d_in[8];
  const float* h_Whh = (const float*)d_in[9];
  const float* h_bih = (const float*)d_in[10];
  const float* h_bhh = (const float*)d_in[11];
  const float* out_W = (const float*)d_in[12];  // [V, H]
  const float* out_b = (const float*)d_in[13];
  float* out = (float*)d_out;

  char* ws = (char*)d_ws;
  size_t off = 0;
  auto take = [&](size_t bytes) -> char* {
    char* p = ws + off;
    off = (off + bytes + 255) & ~(size_t)255;
    return p;
  };
  _Float16* outW_h = (_Float16*)take((size_t)Vd * Hd * 2);
  _Float16* Wx_h   = (_Float16*)take((size_t)H3 * Hd * 2);
  _Float16* Wh_h   = (_Float16*)take((size_t)H3 * Hd * 2);
  _Float16* lWhh_h = (_Float16*)take((size_t)H3 * Hd * 2);
  _Float16* hWih_h = (_Float16*)take((size_t)H3 * Hd * 2);
  _Float16* hWhh_h = (_Float16*)take((size_t)H3 * Hd * 2);
  _Float16* xe_h   = (_Float16*)take((size_t)Bd * Sd * Hd * 2);
  _Float16* hs_h   = (_Float16*)take((size_t)Bd * Sd * Hd * 2);
  float*    gix    = (float*)take((size_t)Bd * Sd * H3 * 4);
  float*    gihh   = (float*)take((size_t)16 * H3 * 4);
  float*    ghl    = (float*)take((size_t)16 * H3 * 4);
  float*    gihi   = (float*)take((size_t)16 * H3 * 4);
  float*    ghhi   = (float*)take((size_t)16 * H3 * 4);
  float*    hl     = (float*)take((size_t)Bd * Hd * 4);
  float*    hh     = (float*)take((size_t)Bd * Hd * 4);
  _Float16* hl_h   = (_Float16*)take((size_t)16 * Hd * 2);
  _Float16* hh_h   = (_Float16*)take((size_t)16 * Hd * 2);

  // ---- one-time prep ----
  init_state<<<64, 256, 0, stream>>>(hl0, hh0, hl, hh, hl_h, hh_h);
  cvt_f16<<<(Vd * Hd + 255) / 256, 256, 0, stream>>>(out_W, outW_h, Vd, Hd, Hd);
  cvt_f16<<<(H3 * Hd + 255) / 256, 256, 0, stream>>>(l_Wih, Wx_h, H3, Hd, 2 * Hd);
  cvt_f16<<<(H3 * Hd + 255) / 256, 256, 0, stream>>>(l_Wih + Hd, Wh_h, H3, Hd, 2 * Hd);
  cvt_f16<<<(H3 * Hd + 255) / 256, 256, 0, stream>>>(l_Whh, lWhh_h, H3, Hd, Hd);
  cvt_f16<<<(H3 * Hd + 255) / 256, 256, 0, stream>>>(h_Wih, hWih_h, H3, Hd, Hd);
  cvt_f16<<<(H3 * Hd + 255) / 256, 256, 0, stream>>>(h_Whh, hWhh_h, H3, Hd, Hd);
  embed_f16<<<(Bd * Sd * Hd) / 256, 256, 0, stream>>>(x, emb, xe_h);

  // gi_x = emb[x] @ Wih_x^T + l_bih (fully parallel over all timesteps)
  gemm_wmma_f16f32<4><<<dim3(H3 / 128, (Bd * Sd) / 64), 256, 0, stream>>>(
      xe_h, Wx_h, l_bih, gix, Bd * Sd, H3, Hd);

  // ---- sequential recurrence: 64 blocks of T=4 steps ----
  for (int blk = 0; blk < Sd / 4; ++blk) {
    // h_h contribution to low-GRU input gates (constant within the block)
    gemm_wmma_f16f32<1><<<dim3(H3 / 128, 1), 256, 0, stream>>>(
        hh_h, Wh_h, nullptr, gihh, 16, H3, Hd);
    for (int t = 0; t < 4; ++t) {
      int s = blk * 4 + t;
      gemm_wmma_f16f32<1><<<dim3(H3 / 128, 1), 256, 0, stream>>>(
          hl_h, lWhh_h, l_bhh, ghl, 16, H3, Hd);
      gru_low_update<<<(Bd * Hd) / 256, 256, 0, stream>>>(
          gix, gihh, ghl, hl, hl_h, hs_h, s);
    }
    // high-level GRU step
    gemm_wmma_f16f32<1><<<dim3(H3 / 128, 1), 256, 0, stream>>>(
        hl_h, hWih_h, h_bih, gihi, 16, H3, Hd);
    gemm_wmma_f16f32<1><<<dim3(H3 / 128, 1), 256, 0, stream>>>(
        hh_h, hWhh_h, h_bhh, ghhi, 16, H3, Hd);
    gru_high_update<<<(Bd * Hd) / 256, 256, 0, stream>>>(gihi, ghhi, hh, hh_h);
  }

  // ---- logits: out[B*S, V] = hs @ out_W^T + out_b (dominant GEMM) ----
  gemm_wmma_f16f32<4><<<dim3(Vd / 128, (Bd * Sd) / 64), 256, 0, stream>>>(
      hs_h, outW_h, out_b, out, Bd * Sd, Vd, Hd);

  // ---- final states ----
  write_states<<<(2 * Bd * Hd) / 256, 256, 0, stream>>>(hl, hh, out);
}
